// GGNNClassifier_7000796692925
// MI455X (gfx1250) — matrix operand order repaired and checked by reading, not compiled, not run on hardware
//
#include <hip/hip_runtime.h>
#include <hip/hip_bf16.h>

// ---------------- problem constants (from reference) ----------------
constexpr int N     = 100000;   // nodes (== 6250 * 16)
constexpr int E     = 1000000;  // edges
constexpr int IN    = 128;
constexpr int HID   = 64;
constexpr int NET   = 4;
constexpr int STEPS = 5;
constexpr int G     = 128;
constexpr int LBL   = 10;
constexpr int NT    = N / 16;   // 6250 node tiles

typedef __attribute__((ext_vector_type(16))) _Float16 v16h;
typedef __attribute__((ext_vector_type(8)))  float    v8f;

// ---------------- WMMA helpers (16x16x32 f16 -> f32) ----------------
// A: 16x32 tile, f16 row-major source.
// Lane m=lane&15, hi=lane>>4 holds A[m][hi*16 .. +16)
// -> contiguous 32-byte run per lane.
__device__ __forceinline__ v16h loadA_f16(const _Float16* __restrict__ p, int ld, int lane) {
  const int m  = lane & 15;
  const int hi = (lane >> 4) & 1;
  const _Float16* r = p + (size_t)m * ld + hi * 16;
  v16h a;
#pragma unroll
  for (int i = 0; i < 8; ++i) a[i] = r[i];
#pragma unroll
  for (int i = 0; i < 8; ++i) a[i + 8] = r[8 + i];
  return a;
}

// B: 32x16 tile (K x Nout) with B[k][n] = W[n][k]; W row-major f16 [out][K].
// Lane n=lane&15, hi=lane>>4 holds W[nbase+n][kbase + hi*16 .. +16) (contiguous 32B).
__device__ __forceinline__ v16h loadB_w(const _Float16* __restrict__ p, int ld, int lane) {
  const int n  = lane & 15;
  const int hi = (lane >> 4) & 1;
  const _Float16* r = p + (size_t)n * ld + hi * 16;
  v16h b;
#pragma unroll
  for (int i = 0; i < 16; ++i) b[i] = r[i];
  return b;
}

// D/C f32 16x16: element v of v8f -> row hi*8+v, col lane&15
__device__ __forceinline__ void storeD_bias_f32(float* __restrict__ p, int ld, int lane,
                                                const v8f& acc, const float* __restrict__ bias) {
  const int n  = lane & 15;
  const int hi = (lane >> 4) & 1;
  const float b = bias[n];
#pragma unroll
  for (int v = 0; v < 8; ++v) p[(size_t)(hi * 8 + v) * ld + n] = acc[v] + b;
}

__device__ __forceinline__ void storeD_bias_f16(_Float16* __restrict__ p, int ld, int lane,
                                                const v8f& acc, const float* __restrict__ bias) {
  const int n  = lane & 15;
  const int hi = (lane >> 4) & 1;
  const float b = bias[n];
#pragma unroll
  for (int v = 0; v < 8; ++v) p[(size_t)(hi * 8 + v) * ld + n] = (_Float16)(acc[v] + b);
}

#define WMMA_F16(A, B, C) \
  __builtin_amdgcn_wmma_f32_16x16x32_f16(false, (A), false, (B), (short)0, (C), false, false)

// ---------------- utility kernels ----------------
__global__ void k_f32_to_f16(const float* __restrict__ s, _Float16* __restrict__ d, int n) {
  int i = blockIdx.x * blockDim.x + threadIdx.x;
  if (i < n) d[i] = (_Float16)s[i];
}

__global__ void k_zero(float* __restrict__ p, int n) {
  int i = blockIdx.x * blockDim.x + threadIdx.x;
  if (i < n) p[i] = 0.0f;
}

// ---------------- GEMM kernels ----------------
// h = x @ W_in^T + b_in ; writes f32 h and f16 mirror. One wave per (node_tile, out_tile).
__global__ void k_linear_in(const _Float16* __restrict__ xH, const _Float16* __restrict__ WinH,
                            const float* __restrict__ b_in, float* __restrict__ h,
                            _Float16* __restrict__ hH) {
  const int wid  = (blockIdx.x * blockDim.x + threadIdx.x) >> 5;
  const int lane = threadIdx.x & 31;
  if (wid >= NT * 4) return;                    // wave-uniform guard
  const int tile = wid >> 2, j = wid & 3;
  const int row0 = tile * 16;
  v8f acc = {};
#pragma unroll
  for (int kc = 0; kc < 4; ++kc) {
    v16h A = loadA_f16(xH + (size_t)row0 * IN + kc * 32, IN, lane);
    v16h B = loadB_w(WinH + (size_t)(j * 16) * IN + kc * 32, IN, lane);
    acc = WMMA_F16(A, B, acc);
  }
  const int n  = lane & 15;
  const int hi = (lane >> 4) & 1;
  const float b = b_in[j * 16 + n];
#pragma unroll
  for (int v = 0; v < 8; ++v) {
    const size_t idx = (size_t)(row0 + hi * 8 + v) * HID + j * 16 + n;
    const float val = acc[v] + b;
    h[idx]  = val;
    hH[idx] = (_Float16)val;
  }
}

// trans[et] = h @ We[et]^T + be[et] (stored f16) ; one wave per (node_tile, etype)
__global__ void k_etype_linear(const _Float16* __restrict__ hH, const _Float16* __restrict__ WeH,
                               const float* __restrict__ be, _Float16* __restrict__ transH) {
  const int wid  = (blockIdx.x * blockDim.x + threadIdx.x) >> 5;
  const int lane = threadIdx.x & 31;
  if (wid >= NT * NET) return;
  const int tile = wid >> 2, et = wid & 3;
  const int row0 = tile * 16;
  v16h A0 = loadA_f16(hH + (size_t)row0 * HID, HID, lane);
  v16h A1 = loadA_f16(hH + (size_t)row0 * HID + 32, HID, lane);
  const _Float16* W = WeH + (size_t)et * HID * HID;
  _Float16* out = transH + (size_t)et * N * HID + (size_t)row0 * HID;
#pragma unroll
  for (int j = 0; j < 4; ++j) {
    v8f acc = {};
    acc = WMMA_F16(A0, loadB_w(W + (size_t)(j * 16) * HID, HID, lane), acc);
    acc = WMMA_F16(A1, loadB_w(W + (size_t)(j * 16) * HID + 32, HID, lane), acc);
    storeD_bias_f16(out + j * 16, HID, lane, acc, be + et * HID + j * 16);
  }
}

// a[dst[e]] += trans[etype[e]][src[e]] ; one wave per edge; scalar index loads
__global__ void k_scatter(const _Float16* __restrict__ transH, const int* __restrict__ src,
                          const int* __restrict__ dst, const int* __restrict__ et,
                          float* __restrict__ a) {
  const int wid = (blockIdx.x * blockDim.x + threadIdx.x) >> 5;
  if (wid >= E) return;                         // wave-uniform guard
  const int e = __builtin_amdgcn_readfirstlane(wid);
  const int lane = threadIdx.x & 31;
  const int s_et = et[e], s_src = src[e], s_dst = dst[e];  // scalar (s_load) candidates
  const _Float16* s = transH + ((size_t)s_et * N + (size_t)s_src) * HID;
  float* d = a + (size_t)s_dst * HID;
  atomicAdd(d + lane,      (float)s[lane]);
  atomicAdd(d + lane + 32, (float)s[lane + 32]);
}

// GRUCell(a, hcur) -> hnext (f32 + f16 mirror) ; one wave per (node_tile, col_group j)
__global__ void k_gru(const _Float16* __restrict__ aH, const float* __restrict__ hcur,
                      const _Float16* __restrict__ hcurH,
                      const _Float16* __restrict__ WihH, const float* __restrict__ b_ih,
                      const _Float16* __restrict__ WhhH, const float* __restrict__ b_hh,
                      float* __restrict__ hnext, _Float16* __restrict__ hnextH) {
  const int wid  = (blockIdx.x * blockDim.x + threadIdx.x) >> 5;
  const int lane = threadIdx.x & 31;
  if (wid >= NT * 4) return;
  const int tile = wid >> 2, j = wid & 3;
  const int row0 = tile * 16;
  const int n  = lane & 15;
  const int hi = (lane >> 4) & 1;

  v16h Aa0 = loadA_f16(aH + (size_t)row0 * HID, HID, lane);
  v16h Aa1 = loadA_f16(aH + (size_t)row0 * HID + 32, HID, lane);
  v16h Ah0 = loadA_f16(hcurH + (size_t)row0 * HID, HID, lane);
  v16h Ah1 = loadA_f16(hcurH + (size_t)row0 * HID + 32, HID, lane);

  v8f ir = {}, iz = {}, in_ = {}, hr = {}, hz = {}, hn = {};
  // gi = a @ W_ih^T  (rows: [r:0..63 | z:64..127 | n:128..191])
  ir  = WMMA_F16(Aa0, loadB_w(WihH + (size_t)(j * 16) * HID,            HID, lane), ir);
  ir  = WMMA_F16(Aa1, loadB_w(WihH + (size_t)(j * 16) * HID + 32,       HID, lane), ir);
  iz  = WMMA_F16(Aa0, loadB_w(WihH + (size_t)(64 + j * 16) * HID,       HID, lane), iz);
  iz  = WMMA_F16(Aa1, loadB_w(WihH + (size_t)(64 + j * 16) * HID + 32,  HID, lane), iz);
  in_ = WMMA_F16(Aa0, loadB_w(WihH + (size_t)(128 + j * 16) * HID,      HID, lane), in_);
  in_ = WMMA_F16(Aa1, loadB_w(WihH + (size_t)(128 + j * 16) * HID + 32, HID, lane), in_);
  // gh = h @ W_hh^T
  hr  = WMMA_F16(Ah0, loadB_w(WhhH + (size_t)(j * 16) * HID,            HID, lane), hr);
  hr  = WMMA_F16(Ah1, loadB_w(WhhH + (size_t)(j * 16) * HID + 32,       HID, lane), hr);
  hz  = WMMA_F16(Ah0, loadB_w(WhhH + (size_t)(64 + j * 16) * HID,       HID, lane), hz);
  hz  = WMMA_F16(Ah1, loadB_w(WhhH + (size_t)(64 + j * 16) * HID + 32,  HID, lane), hz);
  hn  = WMMA_F16(Ah0, loadB_w(WhhH + (size_t)(128 + j * 16) * HID,      HID, lane), hn);
  hn  = WMMA_F16(Ah1, loadB_w(WhhH + (size_t)(128 + j * 16) * HID + 32, HID, lane), hn);

  const int c = j * 16 + n;
  const float bir = b_ih[c],       bhr = b_hh[c];
  const float biz = b_ih[64 + c],  bhz = b_hh[64 + c];
  const float bin = b_ih[128 + c], bhn = b_hh[128 + c];
#pragma unroll
  for (int v = 0; v < 8; ++v) {
    const size_t idx = (size_t)(row0 + hi * 8 + v) * HID + c;
    const float hold = hcur[idx];
    const float r  = 1.0f / (1.0f + __expf(-(ir[v] + bir + hr[v] + bhr)));
    const float z  = 1.0f / (1.0f + __expf(-(iz[v] + biz + hz[v] + bhz)));
    const float nn = tanhf(in_[v] + bin + r * (hn[v] + bhn));
    const float hv = (1.0f - z) * nn + z * hold;
    hnext[idx]  = hv;
    hnextH[idx] = (_Float16)hv;
  }
}

// ---------------- readout ----------------
__global__ void k_graph_acc(const float* __restrict__ h, const int* __restrict__ n2g,
                            float* __restrict__ hgsum, float* __restrict__ cnt) {
  int i = blockIdx.x * blockDim.x + threadIdx.x;
  if (i >= N * HID) return;
  const int node = i >> 6, c = i & 63;
  const int g = n2g[node];
  atomicAdd(&hgsum[(size_t)g * HID + c], h[i]);
  if (c == 0) atomicAdd(&cnt[g], 1.0f);
}

__global__ void k_classifier(const float* __restrict__ hgsum, const float* __restrict__ cnt,
                             const float* __restrict__ W1, const float* __restrict__ b1,
                             const float* __restrict__ W2, const float* __restrict__ b2,
                             float* __restrict__ out) {
  int g = blockIdx.x * blockDim.x + threadIdx.x;
  if (g >= G) return;
  const float inv = 1.0f / fmaxf(cnt[g], 1.0f);
  float hg[HID];
#pragma unroll
  for (int c = 0; c < HID; ++c) hg[c] = hgsum[(size_t)g * HID + c] * inv;
  float hid[HID / 2];
#pragma unroll 4
  for (int o = 0; o < HID / 2; ++o) {
    float s = b1[o];
    for (int c = 0; c < HID; ++c) s += W1[o * HID + c] * hg[c];
    hid[o] = fmaxf(s, 0.0f);
  }
  for (int o = 0; o < LBL; ++o) {
    float s = b2[o];
    for (int c = 0; c < HID / 2; ++c) s += W2[o * (HID / 2) + c] * hid[c];
    out[(size_t)g * LBL + o] = s;
  }
}

// ---------------- host launch ----------------
extern "C" void kernel_launch(void* const* d_in, const int* in_sizes, int n_in,
                              void* d_out, int out_size, void* d_ws, size_t ws_size,
                              hipStream_t stream) {
  (void)in_sizes; (void)n_in; (void)out_size; (void)ws_size;
  const float* x     = (const float*)d_in[0];
  const int*   src   = (const int*)d_in[1];
  const int*   dst   = (const int*)d_in[2];
  const int*   etype = (const int*)d_in[3];
  const int*   n2g   = (const int*)d_in[4];
  const float* W_in  = (const float*)d_in[5];
  const float* b_in  = (const float*)d_in[6];
  const float* We    = (const float*)d_in[7];
  const float* be    = (const float*)d_in[8];
  const float* W_ih  = (const float*)d_in[9];
  const float* b_ih  = (const float*)d_in[10];
  const float* W_hh  = (const float*)d_in[11];
  const float* b_hh  = (const float*)d_in[12];
  const float* W1    = (const float*)d_in[13];
  const float* b1    = (const float*)d_in[14];
  const float* W2    = (const float*)d_in[15];
  const float* b2    = (const float*)d_in[16];
  float* out = (float*)d_out;

  char* ws = (char*)d_ws;
  size_t off = 0;
  auto alloc = [&](size_t bytes) -> void* {
    void* p = ws + off;
    off += (bytes + 255) & ~(size_t)255;
    return p;
  };
  float*    h0     = (float*)alloc((size_t)N * HID * 4);
  float*    h1     = (float*)alloc((size_t)N * HID * 4);
  _Float16* h0H    = (_Float16*)alloc((size_t)N * HID * 2);
  _Float16* h1H    = (_Float16*)alloc((size_t)N * HID * 2);
  float*    a      = (float*)alloc((size_t)N * HID * 4);
  _Float16* aH     = (_Float16*)alloc((size_t)N * HID * 2);
  _Float16* transH = (_Float16*)alloc((size_t)NET * N * HID * 2);
  _Float16* xH     = (_Float16*)alloc((size_t)N * IN * 2);
  _Float16* WinH   = (_Float16*)alloc((size_t)HID * IN * 2);
  _Float16* WeH    = (_Float16*)alloc((size_t)NET * HID * HID * 2);
  _Float16* WihH   = (_Float16*)alloc((size_t)3 * HID * HID * 2);
  _Float16* WhhH   = (_Float16*)alloc((size_t)3 * HID * HID * 2);
  float*    hgsum  = (float*)alloc((size_t)G * HID * 4);
  float*    cnt    = (float*)alloc((size_t)G * 4);

  const int TPB = 256;
  const int WPB = TPB / 32;
  const dim3 blk(TPB);
  const dim3 gGemm((NT * 4 + WPB - 1) / WPB);   // (tile, j) / (tile, et) waves
  const dim3 gScat((E + WPB - 1) / WPB);        // one wave per edge
  const dim3 gNH((N * HID + TPB - 1) / TPB);

  // f32 -> f16 staging (weights + input features)
  k_f32_to_f16<<<(HID * IN + TPB - 1) / TPB, blk, 0, stream>>>(W_in, WinH, HID * IN);
  k_f32_to_f16<<<(NET * HID * HID + TPB - 1) / TPB, blk, 0, stream>>>(We, WeH, NET * HID * HID);
  k_f32_to_f16<<<(3 * HID * HID + TPB - 1) / TPB, blk, 0, stream>>>(W_ih, WihH, 3 * HID * HID);
  k_f32_to_f16<<<(3 * HID * HID + TPB - 1) / TPB, blk, 0, stream>>>(W_hh, WhhH, 3 * HID * HID);
  k_f32_to_f16<<<(N * IN + TPB - 1) / TPB, blk, 0, stream>>>(x, xH, N * IN);

  // h0 = x @ W_in^T + b_in
  k_linear_in<<<gGemm, blk, 0, stream>>>(xH, WinH, b_in, h0, h0H);

  float*    hcur  = h0;
  float*    hnxt  = h1;
  _Float16* hcurH = h0H;
  _Float16* hnxtH = h1H;
  for (int s = 0; s < STEPS; ++s) {
    k_etype_linear<<<gGemm, blk, 0, stream>>>(hcurH, WeH, be, transH);
    k_zero<<<gNH, blk, 0, stream>>>(a, N * HID);
    k_scatter<<<gScat, blk, 0, stream>>>(transH, src, dst, etype, a);
    k_f32_to_f16<<<gNH, blk, 0, stream>>>(a, aH, N * HID);
    k_gru<<<gGemm, blk, 0, stream>>>(aH, hcur, hcurH, WihH, b_ih, WhhH, b_hh, hnxt, hnxtH);
    float* t = hcur; hcur = hnxt; hnxt = t;
    _Float16* tH = hcurH; hcurH = hnxtH; hnxtH = tH;
  }

  // readout
  k_zero<<<(G * HID + TPB - 1) / TPB, blk, 0, stream>>>(hgsum, G * HID);
  k_zero<<<1, blk, 0, stream>>>(cnt, G);
  k_graph_acc<<<gNH, blk, 0, stream>>>(hcur, n2g, hgsum, cnt);
  k_classifier<<<(G + TPB - 1) / TPB, blk, 0, stream>>>(hgsum, cnt, W1, b1, W2, b2, out);
}
